// LinearAttention__33861522162471
// MI455X (gfx1250) — matrix operand order repaired and checked by reading, not compiled, not run on hardware
//
#include <hip/hip_runtime.h>
#include <hip/hip_bf16.h>

// ---------------------------------------------------------------------------
// CDNA5 (gfx1250) linear attention, all GEMMs via v_wmma_f32_16x16x32_bf16.
// Memory-bound (~73 GFLOP vs ~128MB HBM traffic): intermediates kept in bf16
// (96MB qkv stays L2-resident), double-buffered LDS staging in the two big
// GEMMs, and all epilogue stores coalesced to 128B rows via LDS transpose.
// ---------------------------------------------------------------------------

typedef __attribute__((ext_vector_type(16))) __bf16          v16bf;
typedef __attribute__((ext_vector_type(8)))  float           v8f;
typedef __attribute__((ext_vector_type(4)))  float           v4f;
typedef __attribute__((ext_vector_type(8)))  unsigned short  v8us;

union FragB16 { v16bf v; v8us h[2]; };

__device__ __forceinline__ unsigned short f2bf(float f) {
    union { float f; unsigned u; } x; x.f = f;
    unsigned r = x.u + 0x7FFFu + ((x.u >> 16) & 1u);   // round-to-nearest-even
    return (unsigned short)(r >> 16);
}
__device__ __forceinline__ unsigned pk2(float lo, float hi) {
    return (unsigned)f2bf(lo) | ((unsigned)f2bf(hi) << 16);
}

__device__ __forceinline__ float phi_elu1(float t) {
    return t > 0.0f ? (t + 1.0f) : __expf(t);   // elu(t)+1
}

__device__ __forceinline__ v8f vzero8() {
    v8f z;
#pragma unroll
    for (int i = 0; i < 8; ++i) z[i] = 0.0f;
    return z;
}

__device__ __forceinline__ v8f wmma_bf16(v16bf a, v16bf b, v8f c) {
    return __builtin_amdgcn_wmma_f32_16x16x32_bf16(
        false, a, false, b, (short)0, c, false, false);
}

// 16 f32 -> 8 packed bf16x2 dwords
struct Pack8 { unsigned d[8]; };
__device__ __forceinline__ Pack8 cvt16(const float* __restrict__ src) {
    v4f a0 = *(const v4f*)(src + 0);
    v4f a1 = *(const v4f*)(src + 4);
    v4f a2 = *(const v4f*)(src + 8);
    v4f a3 = *(const v4f*)(src + 12);
    Pack8 p;
    p.d[0] = pk2(a0[0], a0[1]); p.d[1] = pk2(a0[2], a0[3]);
    p.d[2] = pk2(a1[0], a1[1]); p.d[3] = pk2(a1[2], a1[3]);
    p.d[4] = pk2(a2[0], a2[1]); p.d[5] = pk2(a2[2], a2[3]);
    p.d[6] = pk2(a3[0], a3[1]); p.d[7] = pk2(a3[2], a3[3]);
    return p;
}
__device__ __forceinline__ void st8(unsigned short* dst, const Pack8& p) {
    unsigned* d = (unsigned*)dst;
#pragma unroll
    for (int i = 0; i < 8; ++i) d[i] = p.d[i];   // merges to 2x ds_store_b128
}

// WMMA fragment loads from LDS (stride in ushorts; row pitch = 16B multiple)
__device__ __forceinline__ v16bf loadA(const unsigned short* base, int row,
                                       int stride, int kk, int half) {
    FragB16 f;
    const unsigned short* p = base + row * stride + kk + half * 8;
    f.h[0] = *(const v8us*)p;
    f.h[1] = *(const v8us*)(p + 16);
    return f.v;
}
__device__ __forceinline__ v16bf loadB(const unsigned short* base, int row,
                                       int stride, int kk, int half) {
    FragB16 f;
    const unsigned short* p = base + row * stride + kk + half * 8;
    f.h[0] = *(const v8us*)p;
    f.h[1] = *(const v8us*)(p + 16);
    return f.v;
}

// Problem constants
#define BATCH 4
#define SEQ   8192
#define DIMC  512
#define HEADS 8
#define HDIM  64
#define THREEC 1536
#define MROWS (BATCH * SEQ)
#define SCALE 0.125f

#define LDK32 40   // 80B rows  (32-wide K tiles)
#define LDK64 72   // 144B rows (64-wide tiles)

// Ping-pong layout inside a 20480-ushort (40KB) smem block:
//   A buffer t at offset t*10240, B buffer t at offset 5120 + t*10240.
#define PPA(t) ((t) * 10240)
#define PPB(t) (5120 + (t) * 10240)

// ---------------------------------------------------------------------------
// Pass 1: qkv = x @ Wqkv^T, phi on q/k thirds, bf16 out [32768][1536].
// 128x128 block tile, 8 waves = 4(M)x2(N), wave tile 32x64.
// Double-buffered LDS; epilogue staged through LDS -> coalesced 128B stores.
// ---------------------------------------------------------------------------
__global__ __launch_bounds__(256) void qkv_phi_kernel(
    const float* __restrict__ x, const float* __restrict__ w,
    unsigned short* __restrict__ qkv)
{
    __shared__ __attribute__((aligned(16))) unsigned short smem[20480]; // 40KB

    const int tid  = threadIdx.x;
    const int lane = tid & 31;
    const int wave = tid >> 5;
    const int half = lane >> 4;
    const int l15  = lane & 15;
    const int mBase = blockIdx.y * 128;
    const int oBase = blockIdx.x * 128;
    const int wm = (wave & 3) * 32;
    const int wn = (wave >> 2) * 64;

    v8f acc[2][4];
#pragma unroll
    for (int i = 0; i < 2; ++i)
#pragma unroll
        for (int j = 0; j < 4; ++j) acc[i][j] = vzero8();

    const int lrow = tid >> 1;          // 0..127
    const int lcol = (tid & 1) * 16;    // 0 or 16
    const float* xsrc = x + (size_t)(mBase + lrow) * DIMC + lcol;
    const float* wsrc = w + (size_t)(oBase + lrow) * DIMC + lcol;
    const int soff = lrow * LDK32 + lcol;

    // prologue: stage tile 0
    st8(smem + PPA(0) + soff, cvt16(xsrc));
    st8(smem + PPB(0) + soff, cvt16(wsrc));
    __syncthreads();

    for (int t = 0; t < 16; ++t) {
        const int cur = t & 1, nxt = cur ^ 1;
        const unsigned short* Acur = smem + PPA(cur);
        const unsigned short* Bcur = smem + PPB(cur);
        Pack8 pa, pb;
        if (t < 15) {                       // prefetch tile t+1 into regs
            pa = cvt16(xsrc + (t + 1) * 32);
            pb = cvt16(wsrc + (t + 1) * 32);
        }
        FragB16 af[2];
        v16bf   bf[4];
#pragma unroll
        for (int i = 0; i < 2; ++i) af[i].v = loadA(Acur, wm + i*16 + l15, LDK32, 0, half);
#pragma unroll
        for (int j = 0; j < 4; ++j) bf[j]   = loadB(Bcur, wn + j*16 + l15, LDK32, 0, half);
#pragma unroll
        for (int i = 0; i < 2; ++i)
#pragma unroll
            for (int j = 0; j < 4; ++j)
                acc[i][j] = wmma_bf16(af[i].v, bf[j], acc[i][j]);
        if (t < 15) {
            st8(smem + PPA(nxt) + soff, pa);
            st8(smem + PPB(nxt) + soff, pb);
        }
        __syncthreads();
    }

    // ---- epilogue: stage wave tile (32x64 bf16) in LDS, coalesced stores
    unsigned short* ot = smem + wave * 2048;     // [32 rows][64 cols]
#pragma unroll
    for (int j = 0; j < 4; ++j) {
        const bool doPhi = (oBase + wn + j * 16) < 2 * DIMC;   // lane-invariant
#pragma unroll
        for (int i = 0; i < 2; ++i)
#pragma unroll
            for (int r = 0; r < 8; ++r) {
                float v = acc[i][j][r];
                if (doPhi) v = phi_elu1(v);
                ot[(i*16 + r + half*8) * 64 + j*16 + l15] = f2bf(v);
            }
    }
    __syncthreads();
    const unsigned short* srow = ot + lane * 64;
    unsigned short* drow = qkv + (size_t)(mBase + wm + lane) * THREEC + oBase + wn;
#pragma unroll
    for (int q = 0; q < 8; ++q)
        *(v8us*)(drow + q * 8) = *(const v8us*)(srow + q * 8);
}

// ---------------------------------------------------------------------------
__global__ void zero_f32(float* __restrict__ p, int n)
{
    int i = blockIdx.x * 256 + threadIdx.x;
    if (i < n) p[i] = 0.0f;
}

// ---------------------------------------------------------------------------
// Pass 2: kv[b,h,d,e] = sum_n phik[n,d]*v[n,e] * SCALE.
// A staged transposed during global->LDS copy; N split 16 ways with
// global_atomic_add_f32 reduction. Each of 8 waves owns 2 output tiles.
// ---------------------------------------------------------------------------
__global__ __launch_bounds__(256) void kv_state_kernel(
    const unsigned short* __restrict__ qkv, float* __restrict__ kv)
{
    __shared__ __attribute__((aligned(16))) unsigned short Kt[64 * LDK64];
    __shared__ __attribute__((aligned(16))) unsigned short Vt[64 * LDK64];

    const int bh  = blockIdx.x;
    const int seg = blockIdx.y;
    const int b = bh >> 3, h = bh & 7;
    const unsigned short* kp = qkv + (size_t)b * SEQ * THREEC + DIMC   + h * HDIM;
    const unsigned short* vp = qkv + (size_t)b * SEQ * THREEC + 2*DIMC + h * HDIM;

    const int tid = threadIdx.x;
    const int lane = tid & 31, wave = tid >> 5;
    const int half = lane >> 4, l15 = lane & 15;
    const int tm  = wave & 3;
    const int tn0 = (wave >> 2) * 2;

    v8f acc2[2];
    acc2[0] = vzero8(); acc2[1] = vzero8();

    const int srw = tid >> 2;
    const int sc  = (tid & 3) * 16;

    for (int c = 0; c < 8; ++c) {
        const int n0 = seg * 512 + c * 64;
        {
            const unsigned short* kr = kp + (size_t)(n0 + srw) * THREEC + sc;
            const unsigned short* vr = vp + (size_t)(n0 + srw) * THREEC + sc;
            v8us k0 = *(const v8us*)kr, k1 = *(const v8us*)(kr + 8);
            v8us w0 = *(const v8us*)vr, w1 = *(const v8us*)(vr + 8);
#pragma unroll
            for (int e = 0; e < 8; ++e) {
                Kt[(sc + e)     * LDK64 + srw] = k0[e];
                Kt[(sc + 8 + e) * LDK64 + srw] = k1[e];
                Vt[(sc + e)     * LDK64 + srw] = w0[e];
                Vt[(sc + 8 + e) * LDK64 + srw] = w1[e];
            }
        }
        __syncthreads();
#pragma unroll
        for (int kk = 0; kk < 64; kk += 32) {
            v16bf a = loadA(Kt, tm*16 + l15, LDK64, kk, half);
#pragma unroll
            for (int j = 0; j < 2; ++j) {
                v16bf bb = loadB(Vt, (tn0 + j)*16 + l15, LDK64, kk, half);
                acc2[j] = wmma_bf16(a, bb, acc2[j]);
            }
        }
        __syncthreads();
    }

#pragma unroll
    for (int j = 0; j < 2; ++j)
#pragma unroll
        for (int r = 0; r < 8; ++r) {
            const int d = tm*16 + r + half*8;
            const int e = (tn0 + j)*16 + l15;
            atomicAdd(&kv[(size_t)bh * 4096 + d * 64 + e], acc2[j][r] * SCALE);
        }
}

// ---------------------------------------------------------------------------
// Pass 3: attn[b,n,h*64+e] = sum_d phiq[b,n,h,d] * kv[b,h,d,e]  (bf16 out)
// ---------------------------------------------------------------------------
__global__ __launch_bounds__(256) void attn_kernel(
    const unsigned short* __restrict__ qkv, const float* __restrict__ kv,
    unsigned short* __restrict__ attn)
{
    __shared__ __attribute__((aligned(16))) unsigned short Aq[128 * LDK64];
    __shared__ __attribute__((aligned(16))) unsigned short Bt[64  * LDK64];

    const int nb = blockIdx.x;
    const int bh = blockIdx.y;
    const int b = bh >> 3, h = bh & 7;
    const int nBase = nb * 128;
    const unsigned short* qp = qkv + (size_t)b * SEQ * THREEC + h * HDIM;

    const int tid = threadIdx.x;
    const int lane = tid & 31, wave = tid >> 5;
    const int half = lane >> 4, l15 = lane & 15;

    {   // stage A: 128x64 bf16 direct copy
        const int row = tid >> 1;
        const int dc  = (tid & 1) * 32;
        const unsigned short* src = qp + (size_t)(nBase + row) * THREEC + dc;
#pragma unroll
        for (int q = 0; q < 4; ++q)
            *(v8us*)&Aq[row * LDK64 + dc + q*8] = *(const v8us*)(src + q*8);
    }
    {   // stage B^T: kv[d][e] f32 -> Bt[e][d] bf16
        const int d  = tid >> 2;
        const int ec = (tid & 3) * 16;
        const float* src = kv + (size_t)bh * 4096 + d * 64 + ec;
#pragma unroll
        for (int q = 0; q < 4; ++q) {
            v4f t = *(const v4f*)(src + q * 4);
#pragma unroll
            for (int e2 = 0; e2 < 4; ++e2)
                Bt[(ec + q*4 + e2) * LDK64 + d] = f2bf(t[e2]);
        }
    }
    __syncthreads();

    v8f acc4[4];
#pragma unroll
    for (int j = 0; j < 4; ++j) acc4[j] = vzero8();

#pragma unroll
    for (int kk = 0; kk < 64; kk += 32) {
        v16bf a = loadA(Aq, wave*16 + l15, LDK64, kk, half);
#pragma unroll
        for (int j = 0; j < 4; ++j) {
            v16bf bb = loadB(Bt, j*16 + l15, LDK64, kk, half);
            acc4[j] = wmma_bf16(a, bb, acc4[j]);
        }
    }
    __syncthreads();

    // staged epilogue: wave tile 16x64 bf16, coalesced 64B row stores
    unsigned short* ot = Aq + wave * 1024;      // [16 rows][64 cols]
#pragma unroll
    for (int j = 0; j < 4; ++j)
#pragma unroll
        for (int r = 0; r < 8; ++r)
            ot[(r + half*8) * 64 + j*16 + l15] = f2bf(acc4[j][r]);
    __syncthreads();
    {
        const int row = lane >> 1;              // 0..15
        const int ch  = (lane & 1) * 32;        // col half
        const unsigned short* srow = ot + row * 64 + ch;
        unsigned short* drow = attn + (size_t)(b * SEQ + nBase + wave*16 + row) * DIMC
                                    + h * HDIM + ch;
#pragma unroll
        for (int q = 0; q < 4; ++q)
            *(v8us*)(drow + q * 8) = *(const v8us*)(srow + q * 8);
    }
}

// ---------------------------------------------------------------------------
// Pass 4: out = attn @ Wproj^T + bias (f32). Double-buffered like pass 1;
// f32 epilogue staged through LDS in two 32-col rounds -> 128B row stores.
// ---------------------------------------------------------------------------
__global__ __launch_bounds__(256) void proj_kernel(
    const unsigned short* __restrict__ a, const float* __restrict__ w,
    const float* __restrict__ bias, float* __restrict__ out)
{
    __shared__ __attribute__((aligned(16))) unsigned short smem[20480]; // 40KB

    const int tid  = threadIdx.x;
    const int lane = tid & 31, wave = tid >> 5;
    const int half = lane >> 4, l15 = lane & 15;
    const int mBase = blockIdx.y * 128;
    const int oBase = blockIdx.x * 128;
    const int wm = (wave & 3) * 32;
    const int wn = (wave >> 2) * 64;

    v8f acc[2][4];
#pragma unroll
    for (int i = 0; i < 2; ++i)
#pragma unroll
        for (int j = 0; j < 4; ++j) acc[i][j] = vzero8();

    const int lrow = tid >> 1;
    const int lcol = (tid & 1) * 16;
    const unsigned short* asrc = a + (size_t)(mBase + lrow) * DIMC + lcol;
    const float* wsrc = w + (size_t)(oBase + lrow) * DIMC + lcol;
    const int soff = lrow * LDK32 + lcol;

    {   // prologue
        *(v8us*)(smem + PPA(0) + soff)     = *(const v8us*)(asrc);
        *(v8us*)(smem + PPA(0) + soff + 8) = *(const v8us*)(asrc + 8);
        st8(smem + PPB(0) + soff, cvt16(wsrc));
    }
    __syncthreads();

    for (int t = 0; t < 16; ++t) {
        const int cur = t & 1, nxt = cur ^ 1;
        const unsigned short* Acur = smem + PPA(cur);
        const unsigned short* Bcur = smem + PPB(cur);
        v8us ra0, ra1; Pack8 pb;
        if (t < 15) {
            ra0 = *(const v8us*)(asrc + (t + 1) * 32);
            ra1 = *(const v8us*)(asrc + (t + 1) * 32 + 8);
            pb  = cvt16(wsrc + (t + 1) * 32);
        }
        FragB16 af[2];
        v16bf   bf[4];
#pragma unroll
        for (int i = 0; i < 2; ++i) af[i].v = loadA(Acur, wm + i*16 + l15, LDK32, 0, half);
#pragma unroll
        for (int j = 0; j < 4; ++j) bf[j]   = loadB(Bcur, wn + j*16 + l15, LDK32, 0, half);
#pragma unroll
        for (int i = 0; i < 2; ++i)
#pragma unroll
            for (int j = 0; j < 4; ++j)
                acc[i][j] = wmma_bf16(af[i].v, bf[j], acc[i][j]);
        if (t < 15) {
            *(v8us*)(smem + PPA(nxt) + soff)     = ra0;
            *(v8us*)(smem + PPA(nxt) + soff + 8) = ra1;
            st8(smem + PPB(nxt) + soff, pb);
        }
        __syncthreads();
    }

    // ---- f32 epilogue staged in LDS: two 32-col rounds, 128B row stores
    float* ot = (float*)smem + wave * 1024;     // [32 rows][32 cols] per wave
#pragma unroll
    for (int p = 0; p < 2; ++p) {
#pragma unroll
        for (int jj = 0; jj < 2; ++jj) {
            const int j = p * 2 + jj;
            const float bv = bias[oBase + wn + j*16 + l15];
#pragma unroll
            for (int i = 0; i < 2; ++i)
#pragma unroll
                for (int r = 0; r < 8; ++r)
                    ot[(i*16 + r + half*8) * 32 + jj*16 + l15] = acc[i][j][r] + bv;
        }
        __syncthreads();
        const float* srow = ot + lane * 32;
        float* drow = out + (size_t)(mBase + wm + lane) * DIMC + oBase + wn + p * 32;
#pragma unroll
        for (int q = 0; q < 8; ++q)
            *(v4f*)(drow + q * 4) = *(const v4f*)(srow + q * 4);
        __syncthreads();
    }
}

// ---------------------------------------------------------------------------
extern "C" void kernel_launch(void* const* d_in, const int* in_sizes, int n_in,
                              void* d_out, int out_size, void* d_ws, size_t ws_size,
                              hipStream_t stream)
{
    const float* x      = (const float*)d_in[0];
    const float* w_qkv  = (const float*)d_in[1];
    const float* w_proj = (const float*)d_in[2];
    const float* b_proj = (const float*)d_in[3];
    float* out = (float*)d_out;

    // Workspace: qkv bf16 (96MB) | attn bf16 (32MB) | kv f32 (512KB)
    unsigned short* qkv_ws  = (unsigned short*)d_ws;
    unsigned short* attn_ws = qkv_ws + (size_t)MROWS * THREEC;
    float*          kv_ws   = (float*)(attn_ws + (size_t)MROWS * DIMC);

    dim3 blk(256);
    qkv_phi_kernel<<<dim3(THREEC / 128, MROWS / 128), blk, 0, stream>>>(x, w_qkv, qkv_ws);
    zero_f32<<<dim3((BATCH * HEADS * HDIM * HDIM + 255) / 256), blk, 0, stream>>>(
        kv_ws, BATCH * HEADS * HDIM * HDIM);
    kv_state_kernel<<<dim3(BATCH * HEADS, 16), blk, 0, stream>>>(qkv_ws, kv_ws);
    attn_kernel<<<dim3(SEQ / 128, BATCH * HEADS), blk, 0, stream>>>(qkv_ws, kv_ws, attn_ws);
    proj_kernel<<<dim3(DIMC / 128, MROWS / 128), blk, 0, stream>>>(attn_ws, w_proj, b_proj, out);
}
